// FusionChannelInteraction_13460427506003
// MI455X (gfx1250) — matrix operand order repaired and checked by reading, not compiled, not run on hardware
//
#include <hip/hip_runtime.h>
#include <hip/hip_bf16.h>
#include <cstdint>

// ---------------------------------------------------------------------------
// CDNA5 (gfx1250) wave32 WMMA implementation of the fused channel-interaction
// block. Heavy GEMMs: v_wmma_f32_16x16x32_f16 with async global->LDS DMA
// (GLOBAL_LOAD_ASYNC_TO_LDS_B128 + ASYNCcnt) and LDS double buffering.
// ---------------------------------------------------------------------------

typedef __attribute__((ext_vector_type(16))) _Float16 v16h;
typedef __attribute__((ext_vector_type(8)))  float    v8f;

#define LSTR 40           // LDS row stride in halfs (32 data + 8 pad)
#define TILEH (64 * LSTR) // halfs per LDS tile buffer
#define BN_EPS 1e-5f

// ---------------- async global->LDS copy primitives ------------------------
__device__ __forceinline__ uint32_t lds_off(const void* p) {
  // generic -> LDS address space; ptrtoint yields byte offset from LDS base
  return (uint32_t)(uintptr_t)(__attribute__((address_space(3))) const char*)p;
}
__device__ __forceinline__ void async_copy16(const _Float16* gsrc, _Float16* ldst) {
  asm volatile("global_load_async_to_lds_b128 %0, %1, off"
               :: "v"(lds_off(ldst)), "v"((unsigned long long)(uintptr_t)gsrc)
               : "memory");
}
__device__ __forceinline__ void async_wait4() {
  asm volatile("s_wait_asynccnt 0x4" ::: "memory");
}
__device__ __forceinline__ void async_wait0() {
  asm volatile("s_wait_asynccnt 0x0" ::: "memory");
}

// 64 rows x 32 halfs tile; 128 threads, 2 chunks (16B) each -> 2 async ops/wave
__device__ __forceinline__ void async_tile(_Float16* dst, const _Float16* __restrict__ src,
                                           long src_stride, int tid) {
#pragma unroll
  for (int it = 0; it < 2; ++it) {
    int c = tid + it * 128;
    int row = c >> 2, ch = (c & 3) << 3;
    async_copy16(src + (long)row * src_stride + ch, dst + row * LSTR + ch);
  }
}
__device__ __forceinline__ void async_tile_clamped(_Float16* dst, const _Float16* __restrict__ src,
                                                   int row0, int maxrow, long src_stride,
                                                   int k0, int tid) {
#pragma unroll
  for (int it = 0; it < 2; ++it) {
    int c = tid + it * 128;
    int row = c >> 2, ch = (c & 3) << 3;
    int gr = row0 + row; if (gr > maxrow) gr = maxrow;
    async_copy16(src + (long)gr * src_stride + k0 + ch, dst + row * LSTR + ch);
  }
}
// conv B-tile from zero-padded zP [256 x 512] (16x16 spatial grid, border = 0).
// K ordered tap*512+ci so a 32-wide K slice stays in one tap; branch-free.
__device__ __forceinline__ void async_conv_btile(_Float16* dst, const _Float16* __restrict__ zPb,
                                                 int n0, int k0, int tid) {
#pragma unroll
  for (int it = 0; it < 2; ++it) {
    int c = tid + it * 128;
    int row = c >> 2, ch = (c & 3) << 3;
    int n = n0 + row; if (n >= 196) n = 0;   // tile-pad lanes: masked in epilogue
    int k = k0 + ch;
    int tap = k >> 9, ci = k & 511;
    int rr = (n / 14 + tap / 3) * 16 + (n % 14 + tap % 3);   // 0..255, border rows zero
    async_copy16(zPb + (long)rr * 512 + ci, dst + row * LSTR + ch);
  }
}

// ---------------- WMMA fragment loaders (wave32 layouts per CDNA5 ISA) -----
__device__ __forceinline__ v16h ldsA_frag(const _Float16* As, int row0, int lane) {
  int m  = lane & 15;
  int kh = (lane >> 4) << 3;
  const _Float16* p = As + (row0 + m) * LSTR + kh;
  v16h a;
#pragma unroll
  for (int i = 0; i < 8; ++i) { a[i] = p[i]; a[i + 8] = p[16 + i]; }
  return a;
}
__device__ __forceinline__ v16h ldsB_frag(const _Float16* Bs, int col0, int lane) {
  int n  = lane & 15;
  int kb = (lane >> 4) << 3;
  const _Float16* p = Bs + (col0 + n) * LSTR + kb;
  v16h b;
#pragma unroll
  for (int i = 0; i < 8; ++i) { b[i] = p[i]; b[i + 8] = p[16 + i]; }
  return b;
}

#define WMMA_STEP(As, Bs, lane, wr, wc, acc)                                           \
  {                                                                                    \
    v16h a0 = ldsA_frag((As), (wr), (lane));                                           \
    v16h a1 = ldsA_frag((As), (wr) + 16, (lane));                                      \
    v16h b0 = ldsB_frag((Bs), (wc), (lane));                                           \
    v16h b1 = ldsB_frag((Bs), (wc) + 16, (lane));                                      \
    acc[0][0] = __builtin_amdgcn_wmma_f32_16x16x32_f16(false, a0, false, b0, (short)0, \
                                                       acc[0][0], false, false);       \
    acc[0][1] = __builtin_amdgcn_wmma_f32_16x16x32_f16(false, a0, false, b1, (short)0, \
                                                       acc[0][1], false, false);       \
    acc[1][0] = __builtin_amdgcn_wmma_f32_16x16x32_f16(false, a1, false, b0, (short)0, \
                                                       acc[1][0], false, false);       \
    acc[1][1] = __builtin_amdgcn_wmma_f32_16x16x32_f16(false, a1, false, b1, (short)0, \
                                                       acc[1][1], false, false);       \
  }

// =========================== small prep kernels ============================
__global__ void zero_f32_kernel(float* __restrict__ p, long n) {
  long i = (long)blockIdx.x * blockDim.x + threadIdx.x;
  if (i < n) p[i] = 0.f;
}
__global__ void cvt_f16_kernel(const float* __restrict__ in, _Float16* __restrict__ out, int n) {
  int i = blockIdx.x * blockDim.x + threadIdx.x;
  if (i < n) out[i] = (_Float16)in[i];
}
// OIHW [512][512][3][3] f32 -> [co][tap*512+ci] f16
__global__ void reorder_w_kernel(const float* __restrict__ w, _Float16* __restrict__ out) {
  int i = blockIdx.x * blockDim.x + threadIdx.x;
  if (i >= 512 * 512 * 9) return;
  int co = i / (512 * 9);
  int rem = i - co * 512 * 9;
  int ci = rem / 9, t = rem - ci * 9;
  out[(long)co * 4608 + t * 512 + ci] = (_Float16)w[i];
}
// x [128][2048][196] f32 -> Xt [128][208][2048] f16 (pad rows masked downstream)
__global__ void transpose_x_kernel(const float* __restrict__ x, _Float16* __restrict__ Xt) {
  __shared__ float tile[32][33];
  int b = blockIdx.z;
  int k0 = blockIdx.x * 32;
  int n0 = blockIdx.y * 32;
  int tx = threadIdx.x, ty = threadIdx.y;
  const float* xb = x + (long)b * 2048 * 196;
  for (int i = ty; i < 32; i += 8) {
    int n = n0 + tx;
    tile[i][tx] = (n < 196) ? xb[(long)(k0 + i) * 196 + n] : 0.f;
  }
  __syncthreads();
  _Float16* xtb = Xt + (long)b * 208 * 2048;
  for (int i = ty; i < 32; i += 8) {
    int n = n0 + i;
    if (n < 196) xtb[(long)n * 2048 + (k0 + tx)] = (_Float16)tile[tx][i];
  }
}

// ============ GEMM 1: reduce conv  F = Wred(512x2048) @ X(2048x196) ========
__global__ void __launch_bounds__(128)
gemm_reduce_kernel(const _Float16* __restrict__ Wh, const _Float16* __restrict__ Xt,
                   const float* __restrict__ g, const float* __restrict__ bb,
                   const float* __restrict__ mm, const float* __restrict__ vv,
                   _Float16* __restrict__ xf, _Float16* __restrict__ xfT) {
  __shared__ _Float16 As[2][TILEH], Bs[2][TILEH];
  int b = blockIdx.z;
  int m0 = blockIdx.y * 64, n0 = blockIdx.x * 64;
  int tid = threadIdx.x, lane = tid & 31, w = tid >> 5;
  int wr = (w >> 1) * 32, wc = (w & 1) * 32;
  const _Float16* Ag = Wh + (long)m0 * 2048;
  const _Float16* Bt = Xt + (long)b * 208 * 2048;
  v8f acc[2][2] = {};
  const int nk = 2048 / 32;
  async_tile(As[0], Ag, 2048, tid);
  async_tile_clamped(Bs[0], Bt, n0, 207, 2048, 0, tid);
  for (int it = 0; it < nk; ++it) {
    int cur = it & 1, nxt = cur ^ 1;
    __syncthreads();
    if (it + 1 < nk) {
      int k0 = (it + 1) * 32;
      async_tile(As[nxt], Ag + k0, 2048, tid);
      async_tile_clamped(Bs[nxt], Bt, n0, 207, 2048, k0, tid);
      async_wait4();
    } else {
      async_wait0();
    }
    __syncthreads();
    WMMA_STEP(As[cur], Bs[cur], lane, wr, wc, acc);
  }
  int kh = lane >> 4, nn = lane & 15;
#pragma unroll
  for (int ti = 0; ti < 2; ++ti)
#pragma unroll
    for (int r = 0; r < 8; ++r) {
      int c = m0 + wr + ti * 16 + r + 8 * kh;
      float sv = g[c] * rsqrtf(vv[c] + BN_EPS);
      float bias = bb[c] - mm[c] * sv;
#pragma unroll
      for (int tj = 0; tj < 2; ++tj) {
        int n = n0 + wc + tj * 16 + nn;
        float val = fmaxf(acc[ti][tj][r] * sv + bias, 0.f);
        if (n >= 196) val = 0.f;                       // exact zero padding
        if (n < 224) xf[((long)b * 512 + c) * 224 + n] = (_Float16)val;
        if (n < 208) xfT[((long)b * 208 + n) * 512 + c] = (_Float16)val;
      }
    }
}

// ============ GEMM 2: Gram  y = xf(512x224) @ xf^T  ->  f32 [512x512] ======
__global__ void __launch_bounds__(128)
gemm_gram_kernel(const _Float16* __restrict__ xf, float* __restrict__ y) {
  __shared__ _Float16 As[2][TILEH], Bs[2][TILEH];
  int b = blockIdx.z;
  int m0 = blockIdx.y * 64, n0 = blockIdx.x * 64;
  int tid = threadIdx.x, lane = tid & 31, w = tid >> 5;
  int wr = (w >> 1) * 32, wc = (w & 1) * 32;
  const _Float16* xb = xf + (long)b * 512 * 224;
  v8f acc[2][2] = {};
  const int nk = 224 / 32;
  async_tile(As[0], xb + (long)m0 * 224, 224, tid);
  async_tile(Bs[0], xb + (long)n0 * 224, 224, tid);
  for (int it = 0; it < nk; ++it) {
    int cur = it & 1, nxt = cur ^ 1;
    __syncthreads();
    if (it + 1 < nk) {
      int k0 = (it + 1) * 32;
      async_tile(As[nxt], xb + (long)m0 * 224 + k0, 224, tid);
      async_tile(Bs[nxt], xb + (long)n0 * 224 + k0, 224, tid);
      async_wait4();
    } else {
      async_wait0();
    }
    __syncthreads();
    WMMA_STEP(As[cur], Bs[cur], lane, wr, wc, acc);
  }
  float* yb = y + (long)b * 512 * 512;
  int kh = lane >> 4, nn = lane & 15;
#pragma unroll
  for (int ti = 0; ti < 2; ++ti)
#pragma unroll
    for (int tj = 0; tj < 2; ++tj)
#pragma unroll
      for (int r = 0; r < 8; ++r)
        yb[(long)(m0 + wr + ti * 16 + r + 8 * kh) * 512 + (n0 + wc + tj * 16 + nn)] =
            acc[ti][tj][r];
}

// ===== dual softmax: P = softmax(y,dim=2), Pm = softmax(-y,dim=2), f16 =====
__global__ void __launch_bounds__(256)
softmax_pair_kernel(const float* __restrict__ y, _Float16* __restrict__ P,
                    _Float16* __restrict__ Pm) {
  int wid = threadIdx.x >> 5, lane = threadIdx.x & 31;
  int row = blockIdx.x * 8 + wid;
  int b = blockIdx.y;
  const float* r = y + ((long)b * 512 + row) * 512;
  float v[16], vmax = -3.0e38f, vmin = 3.0e38f;
#pragma unroll
  for (int i = 0; i < 16; ++i) {
    v[i] = r[lane + 32 * i];
    vmax = fmaxf(vmax, v[i]); vmin = fminf(vmin, v[i]);
  }
#pragma unroll
  for (int m = 16; m >= 1; m >>= 1) {
    vmax = fmaxf(vmax, __shfl_xor(vmax, m, 32));
    vmin = fminf(vmin, __shfl_xor(vmin, m, 32));
  }
  float ep[16], em[16], sp = 0.f, sm = 0.f;
#pragma unroll
  for (int i = 0; i < 16; ++i) {
    ep[i] = expf(v[i] - vmax); em[i] = expf(vmin - v[i]);  // softmax(-y) via min
    sp += ep[i]; sm += em[i];
  }
#pragma unroll
  for (int m = 16; m >= 1; m >>= 1) {
    sp += __shfl_xor(sp, m, 32); sm += __shfl_xor(sm, m, 32);
  }
  float rp = 1.f / sp, rm = 1.f / sm;
  _Float16* pr  = P  + ((long)b * 512 + row) * 512;
  _Float16* pmr = Pm + ((long)b * 512 + row) * 512;
#pragma unroll
  for (int i = 0; i < 16; ++i) {
    pr[lane + 32 * i]  = (_Float16)(ep[i] * rp);
    pmr[lane + 32 * i] = (_Float16)(em[i] * rm);
  }
}

// ===== GEMM 3: z = P(512x512) @ xf -> zero-padded zP [256 x 512] f16 =======
__global__ void __launch_bounds__(128)
gemm_attn_kernel(const _Float16* __restrict__ P, const _Float16* __restrict__ xfT,
                 _Float16* __restrict__ zP) {
  __shared__ _Float16 As[2][TILEH], Bs[2][TILEH];
  int b = blockIdx.z;
  int m0 = blockIdx.y * 64, n0 = blockIdx.x * 64;
  int tid = threadIdx.x, lane = tid & 31, w = tid >> 5;
  int wr = (w >> 1) * 32, wc = (w & 1) * 32;
  const _Float16* Ag = P   + (long)b * 512 * 512 + (long)m0 * 512;
  const _Float16* Bt = xfT + (long)b * 208 * 512;
  v8f acc[2][2] = {};
  const int nk = 512 / 32;
  async_tile(As[0], Ag, 512, tid);
  async_tile_clamped(Bs[0], Bt, n0, 207, 512, 0, tid);
  for (int it = 0; it < nk; ++it) {
    int cur = it & 1, nxt = cur ^ 1;
    __syncthreads();
    if (it + 1 < nk) {
      int k0 = (it + 1) * 32;
      async_tile(As[nxt], Ag + k0, 512, tid);
      async_tile_clamped(Bs[nxt], Bt, n0, 207, 512, k0, tid);
      async_wait4();
    } else {
      async_wait0();
    }
    __syncthreads();
    WMMA_STEP(As[cur], Bs[cur], lane, wr, wc, acc);
  }
  _Float16* zb = zP + (long)b * 256 * 512;
  int kh = lane >> 4, nn = lane & 15;
#pragma unroll
  for (int ti = 0; ti < 2; ++ti)
#pragma unroll
    for (int tj = 0; tj < 2; ++tj)
#pragma unroll
      for (int r = 0; r < 8; ++r) {
        int n = n0 + wc + tj * 16 + nn;
        int c = m0 + wr + ti * 16 + r + 8 * kh;
        if (n < 196) {
          int rr = (n / 14 + 1) * 16 + (n % 14 + 1);   // interior of 16x16 grid
          zb[(long)rr * 512 + c] = (_Float16)acc[ti][tj][r];
        }
      }
}

// ===== GEMM 4: implicit 3x3 conv + BN + ReLU + mean-pool (atomicAdd) =======
__global__ void __launch_bounds__(128)
gemm_conv_pool_kernel(const _Float16* __restrict__ Wr, const _Float16* __restrict__ zP,
                      const float* __restrict__ g, const float* __restrict__ bb,
                      const float* __restrict__ mm, const float* __restrict__ vv,
                      float* __restrict__ pooled) {
  __shared__ _Float16 As[2][TILEH], Bs[2][TILEH];
  int b = blockIdx.z;
  int m0 = blockIdx.y * 64, n0 = blockIdx.x * 64;
  int tid = threadIdx.x, lane = tid & 31, w = tid >> 5;
  int wr = (w >> 1) * 32, wc = (w & 1) * 32;
  const _Float16* Ag = Wr + (long)m0 * 4608;
  const _Float16* zb = zP + (long)b * 256 * 512;
  v8f acc[2][2] = {};
  const int nk = 4608 / 32;
  async_tile(As[0], Ag, 4608, tid);
  async_conv_btile(Bs[0], zb, n0, 0, tid);
  for (int it = 0; it < nk; ++it) {
    int cur = it & 1, nxt = cur ^ 1;
    __syncthreads();
    if (it + 1 < nk) {
      int k0 = (it + 1) * 32;
      async_tile(As[nxt], Ag + k0, 4608, tid);
      async_conv_btile(Bs[nxt], zb, n0, k0, tid);
      async_wait4();
    } else {
      async_wait0();
    }
    __syncthreads();
    WMMA_STEP(As[cur], Bs[cur], lane, wr, wc, acc);
  }
  const float inv196 = 1.f / 196.f;
  int kh = lane >> 4, nn = lane & 15;
#pragma unroll
  for (int ti = 0; ti < 2; ++ti) {
    float rowsum[8];
#pragma unroll
    for (int r = 0; r < 8; ++r) {
      int c = m0 + wr + ti * 16 + r + 8 * kh;
      float sv = g[c] * rsqrtf(vv[c] + BN_EPS);
      float bias = bb[c] - mm[c] * sv;
      float s = 0.f;
#pragma unroll
      for (int tj = 0; tj < 2; ++tj) {
        int n = n0 + wc + tj * 16 + nn;
        float val = fmaxf(acc[ti][tj][r] * sv + bias, 0.f);
        if (n >= 196) val = 0.f;
        s += val;
      }
      s += __shfl_xor(s, 1, 32); s += __shfl_xor(s, 2, 32);
      s += __shfl_xor(s, 4, 32); s += __shfl_xor(s, 8, 32);
      rowsum[r] = s;
    }
    if (nn == 0) {
#pragma unroll
      for (int r = 0; r < 8; ++r) {
        int c = m0 + wr + ti * 16 + r + 8 * kh;
        atomicAdd(pooled + (long)b * 512 + c, rowsum[r] * inv196);
      }
    }
  }
}

// ===================== tiny FC tail (f32 VALU, negligible) =================
__global__ void fc_kernel(const float* __restrict__ in, int in_stride,
                          const float* __restrict__ w, const float* __restrict__ bias,
                          float* __restrict__ out, int out_stride, int out_off,
                          int K, int O, int do_relu) {
  int idx = blockIdx.x * blockDim.x + threadIdx.x;
  if (idx >= 128 * O) return;
  int b = idx / O, o = idx - b * O;
  float s = bias[o];
  const float* ip = in + (long)b * in_stride;
  const float* wp = w + (long)o * K;
  for (int k = 0; k < K; ++k) s += ip[k] * wp[k];
  if (do_relu) s = fmaxf(s, 0.f);
  out[(long)b * out_stride + out_off + o] = s;
}

// ===========================================================================
extern "C" void kernel_launch(void* const* d_in, const int* in_sizes, int n_in,
                              void* d_out, int out_size, void* d_ws, size_t ws_size,
                              hipStream_t stream) {
  (void)in_sizes; (void)n_in; (void)out_size; (void)ws_size;
  const float* x     = (const float*)d_in[0];
  const float* w_red = (const float*)d_in[1];
  const float* g_r   = (const float*)d_in[2];
  const float* b_r   = (const float*)d_in[3];
  const float* m_r   = (const float*)d_in[4];
  const float* v_r   = (const float*)d_in[5];
  const float* w_tf  = (const float*)d_in[6];
  const float* g_tf  = (const float*)d_in[7];
  const float* b_tf  = (const float*)d_in[8];
  const float* m_tf  = (const float*)d_in[9];
  const float* v_tf  = (const float*)d_in[10];
  const float* w_sf  = (const float*)d_in[11];
  const float* g_sf  = (const float*)d_in[12];
  const float* b_sf  = (const float*)d_in[13];
  const float* m_sf  = (const float*)d_in[14];
  const float* v_sf  = (const float*)d_in[15];
  const float* wp1 = (const float*)d_in[16]; const float* bp1 = (const float*)d_in[17];
  const float* wp2 = (const float*)d_in[18]; const float* bp2 = (const float*)d_in[19];
  const float* wc1 = (const float*)d_in[20]; const float* bc1 = (const float*)d_in[21];
  const float* wc2 = (const float*)d_in[22]; const float* bc2 = (const float*)d_in[23];
  const float* wk1 = (const float*)d_in[24]; const float* bk1 = (const float*)d_in[25];
  const float* wk2 = (const float*)d_in[26]; const float* bk2 = (const float*)d_in[27];
  float* out = (float*)d_out;

  // --- bump allocator over workspace ---
  char* p = (char*)d_ws;
  auto alloc = [&](size_t bytes) -> char* {
    char* r = p; p += (bytes + 255) & ~(size_t)255; return r;
  };
  _Float16* Xt    = (_Float16*)alloc((size_t)128 * 208 * 2048 * 2);
  _Float16* Wh    = (_Float16*)alloc((size_t)512 * 2048 * 2);
  _Float16* WrTF  = (_Float16*)alloc((size_t)512 * 4608 * 2);
  _Float16* WrSF  = (_Float16*)alloc((size_t)512 * 4608 * 2);
  _Float16* xf    = (_Float16*)alloc((size_t)128 * 512 * 224 * 2);
  _Float16* xfT   = (_Float16*)alloc((size_t)128 * 208 * 512 * 2);
  float*    y     = (float*)   alloc((size_t)128 * 512 * 512 * 4);
  _Float16* P     = (_Float16*)alloc((size_t)128 * 512 * 512 * 2);
  _Float16* Pm    = (_Float16*)alloc((size_t)128 * 512 * 512 * 2);
  _Float16* zP    = (_Float16*)alloc((size_t)128 * 256 * 512 * 2);  // padded 16x16 grid
  float*    pooled1 = (float*)alloc((size_t)128 * 512 * 4);
  float*    pooled2 = (float*)alloc((size_t)128 * 512 * 4);
  float*    t1  = (float*)alloc((size_t)128 * 200 * 4);
  float*    t2  = (float*)alloc((size_t)128 * 200 * 4);
  float*    cat = (float*)alloc((size_t)128 * 400 * 4);
  float*    t3  = (float*)alloc((size_t)128 * 200 * 4);

  // --- prep ---
  zero_f32_kernel<<<(2 * 128 * 512 + 255) / 256, 256, 0, stream>>>(pooled1, 2 * 128 * 512);
  {
    long nzp = (long)128 * 256 * 512 / 2;      // zP as f32 words
    zero_f32_kernel<<<(int)((nzp + 255) / 256), 256, 0, stream>>>((float*)zP, nzp);
  }
  cvt_f16_kernel<<<(512 * 2048 + 255) / 256, 256, 0, stream>>>(w_red, Wh, 512 * 2048);
  reorder_w_kernel<<<(512 * 512 * 9 + 255) / 256, 256, 0, stream>>>(w_tf, WrTF);
  reorder_w_kernel<<<(512 * 512 * 9 + 255) / 256, 256, 0, stream>>>(w_sf, WrSF);
  transpose_x_kernel<<<dim3(64, 7, 128), dim3(32, 8), 0, stream>>>(x, Xt);

  // --- stage 1: reduce conv + BN + ReLU -> xf (row + transposed, f16) ---
  gemm_reduce_kernel<<<dim3(4, 8, 128), 128, 0, stream>>>(Wh, Xt, g_r, b_r, m_r, v_r, xf, xfT);

  // --- stage 2: Gram matrix y = xf @ xf^T (f32) ---
  gemm_gram_kernel<<<dim3(8, 8, 128), 128, 0, stream>>>(xf, y);

  // --- stage 3: dual softmax -> P, Pm (f16) ---
  softmax_pair_kernel<<<dim3(64, 128), 256, 0, stream>>>(y, P, Pm);

  // --- branch 1: z = P @ xf ; 3x3 conv + BN + ReLU + mean-pool ---
  gemm_attn_kernel<<<dim3(4, 8, 128), 128, 0, stream>>>(P, xfT, zP);
  gemm_conv_pool_kernel<<<dim3(4, 8, 128), 128, 0, stream>>>(WrTF, zP, g_tf, b_tf, m_tf, v_tf,
                                                             pooled1);
  // --- branch 2: z2 = softmax(-y) @ xf ; conv + pool (zP interior rewritten) ---
  gemm_attn_kernel<<<dim3(4, 8, 128), 128, 0, stream>>>(Pm, xfT, zP);
  gemm_conv_pool_kernel<<<dim3(4, 8, 128), 128, 0, stream>>>(WrSF, zP, g_sf, b_sf, m_sf, v_sf,
                                                             pooled2);

  // --- FC tail ---
  int fcgrid = (128 * 200 + 255) / 256;
  fc_kernel<<<fcgrid, 256, 0, stream>>>(pooled1, 512, wp1, bp1, t1, 200, 0, 512, 200, 1);
  fc_kernel<<<fcgrid, 256, 0, stream>>>(t1, 200, wp2, bp2, cat, 400, 0, 200, 200, 0);
  fc_kernel<<<fcgrid, 256, 0, stream>>>(pooled2, 512, wc1, bc1, t2, 200, 0, 512, 200, 1);
  fc_kernel<<<fcgrid, 256, 0, stream>>>(t2, 200, wc2, bc2, cat, 400, 200, 200, 200, 0);
  fc_kernel<<<fcgrid, 256, 0, stream>>>(cat, 400, wk1, bk1, t3, 200, 0, 400, 200, 1);
  fc_kernel<<<fcgrid, 256, 0, stream>>>(t3, 200, wk2, bk2, out, 200, 0, 200, 200, 0);
}
